// DilatedSparseRnnStack_6786048328287
// MI455X (gfx1250) — matrix operand order repaired and checked by reading, not compiled, not run on hardware
//
#include <hip/hip_runtime.h>
#include <hip/hip_bf16.h>

typedef __bf16 bf16;
typedef bf16  v16bf __attribute__((ext_vector_type(16)));
typedef float v8f   __attribute__((ext_vector_type(8)));
typedef float f32x4 __attribute__((ext_vector_type(4)));
typedef unsigned int u32x4 __attribute__((ext_vector_type(4)));

union AB16 { v16bf v; u32x4 u[2]; };
union H4   { bf16 h[4]; unsigned long long q; };

#define XH_STRIDE 456   // max fan (448) + 8 pad -> conflict-free LDS b128, 16B-aligned rows

__device__ __forceinline__ float sigm(float x)      { return 1.0f / (1.0f + __expf(-x)); }
__device__ __forceinline__ float tanh_fast(float x) { return 1.0f - 2.0f / (__expf(2.0f * x) + 1.0f); }

struct Params {
    const float* x;
    const float* b[4];
    const float* Wout;
    const float* bout;
    const bf16*  Wpk[4];
    bf16*        bufH[4];
    float*       bufC[4];
    float*       out;
};

// Pack W (1024 x fan, f32 row-major) into bf16 blocks of 512 elements, one block per
// (N-tile, k-step), in the B-operand lane order this GEMM uses (contiguous k, same as A):
//   block elem = lane*16 + e, value = W[tile*16 + (lane&15)][ks*32 + (lane>>4)*16 + e]
extern "C" __global__ void dsrnn_pack(const float* __restrict__ W, bf16* __restrict__ Wpk,
                                      int fan, int nk)
{
    int total = 64 * nk * 512;  // == 1024*fan
    for (int i = blockIdx.x * blockDim.x + threadIdx.x; i < total; i += gridDim.x * blockDim.x) {
        int e    = i & 15;
        int lane = (i >> 4) & 31;
        int blk  = i >> 9;
        int ks   = blk % nk;
        int tile = blk / nk;
        int n = tile * 16 + (lane & 15);
        int k = ks * 32 + (lane >> 4) * 16 + e;
        Wpk[i] = (bf16)W[(size_t)n * fan + k];
    }
}

// One S2Cell layer, one timestep, 16 batch rows, 256 threads (8 waves).
// Wave w owns gate columns [32w, 32w+32) of each of the 4 gates.
// C-state lives entirely in bufC: pc = bufC[(t-1)%DIL], nc -> bufC[t%DIL].
template<int LI, int FAN, int DIL>
__device__ __forceinline__ void layer_step(
    int t, int w, int lane, int brow0,
    const bf16* __restrict__ Wpk, const float* __restrict__ biasL,
    bf16* __restrict__ bufH, float* __restrict__ bufC,
    bf16* __restrict__ xh, bf16* __restrict__ hcur_l, float* __restrict__ outF)
{
    const int tid  = threadIdx.x;
    const int half = lane >> 4;
    const int idx  = t % DIL;                       // slot to write (and dC slot to read)
    const int idxp = (t == 0) ? 0 : ((t - 1) % DIL); // prevC slot (t=0: dead read)

    // ---- assemble prevH / dH regions of xh: one b128 per thread per region ----
    {
        const int m = tid >> 4, ch = (tid & 15) * 8;
        *(u32x4*)(xh + m * XH_STRIDE + (FAN - 256) + ch) =
            *(const u32x4*)(hcur_l + m * 128 + ch);                     // prevH
        if (DIL > 1 && t >= DIL) {
            const bf16* src = bufH + ((size_t)idx * 1024 + brow0) * 128;
            *(u32x4*)(xh + m * XH_STRIDE + (FAN - 128) + ch) =
                *(const u32x4*)(src + (size_t)m * 128 + ch);            // dH = bufH[idx]
        } else {
            *(u32x4*)(xh + m * XH_STRIDE + (FAN - 128) + ch) =
                *(const u32x4*)(hcur_l + m * 128 + ch);                 // dH = prevH
        }
    }
    __syncthreads();

    // ---- GEMM: gates(16 x 1024) = xh(16 x FAN) @ W^T ----
    constexpr int NK = FAN / 32;
    v8f acc[4][2];
    v8f zero = {0.f, 0.f, 0.f, 0.f, 0.f, 0.f, 0.f, 0.f};
    #pragma unroll
    for (int j = 0; j < 4; ++j) { acc[j][0] = zero; acc[j][1] = zero; }

    const bf16* xrow  = xh + (lane & 15) * XH_STRIDE + half * 16;
    const bf16* wbase = Wpk + lane * 16;
    #pragma unroll 2
    for (int ks = 0; ks < NK; ++ks) {
        AB16 a;
        a.u[0] = *(const u32x4*)(xrow + ks * 32);
        a.u[1] = *(const u32x4*)(xrow + ks * 32 + 8);
        AB16 bm[8];
        #pragma unroll
        for (int j = 0; j < 4; ++j) {
            #pragma unroll
            for (int tt = 0; tt < 2; ++tt) {
                const bf16* bp = wbase + (((size_t)((j * 16 + 2 * w + tt) * NK + ks)) << 9);
                bm[j * 2 + tt].u[0] = *(const u32x4*)(bp);
                bm[j * 2 + tt].u[1] = *(const u32x4*)(bp + 8);
            }
        }
        #pragma unroll
        for (int j = 0; j < 4; ++j) {
            #pragma unroll
            for (int tt = 0; tt < 2; ++tt) {
                acc[j][tt] = __builtin_amdgcn_wmma_f32_16x16x32_bf16(
                    false, a.v, false, bm[j * 2 + tt].v, (short)0, acc[j][tt], false, false);
            }
        }
    }
    __syncthreads();   // all waves done with xh before out-writes below

    // ---- element-wise cell update, in-wave ----
    const int nlo = lane & 15;
    float bb[4][2];
    #pragma unroll
    for (int j = 0; j < 4; ++j)
        #pragma unroll
        for (int tt = 0; tt < 2; ++tt)
            bb[j][tt] = biasL[j * 256 + w * 32 + tt * 16 + nlo];

    #pragma unroll
    for (int tt = 0; tt < 2; ++tt) {
        const int c = w * 32 + tt * 16 + nlo;               // column within a gate
        float* cw = bufC + ((size_t)idx  * 1024 + brow0) * 256 + c;   // write / dC slot
        const float* cp = bufC + ((size_t)idxp * 1024 + brow0) * 256 + c;   // prevC slot
        #pragma unroll
        for (int r = 0; r < 8; ++r) {
            const int m = r + 8 * half;                     // batch row within tile
            float g0 = acc[0][tt][r] + bb[0][tt];
            float g1 = acc[1][tt][r] + bb[1][tt];
            float g2 = acc[2][tt][r] + bb[2][tt];
            float g3 = acc[3][tt][r] + bb[3][tt];
            float fg = sigm(g0 + 1.0f);
            float cd = tanh_fast(g1);
            float al = sigm(g2);
            float og = sigm(g3);
            float pc = cp[(size_t)m * 256];                 // C_{t-1} (t=0: unused garbage)
            float wc = pc;                                  // DIL==1: dC==prevC -> wc==pc
            if (DIL > 1 && t >= DIL) {
                float dc = cw[(size_t)m * 256];             // C_{t-DIL}
                wc = al * pc + (1.0f - al) * dc;
            }
            float nc = (t >= 1) ? (fg * wc + (1.0f - fg) * cd) : cd;
            cw[(size_t)m * 256] = nc;
            float wh = og * nc;
            if (w < 4) {                                    // c < 128 : out_t
                if (LI == 3) outF[m * 128 + c] = wh;
                else         xh[m * XH_STRIDE + c] = (bf16)wh;  // next layer's inp region
            } else {                                        // c >= 128 : h
                const int ch = c - 128;
                bf16 hb = (bf16)wh;
                hcur_l[m * 128 + ch] = hb;
                if (DIL > 1)
                    bufH[((size_t)idx * 1024 + brow0 + m) * 128 + ch] = hb;
            }
        }
    }
    __threadfence_block();
    __syncthreads();
}

extern "C" __global__ void __launch_bounds__(256)
dsrnn_main(Params p)
{
    __shared__ alignas(16) bf16  xh[16 * XH_STRIDE];   // staged activations (bf16)
    __shared__ alignas(16) bf16  hcur[4][16 * 128];    // prevH per layer
    __shared__ alignas(16) float outF[16 * 128];       // layer-3 out (f32)
    __shared__ alignas(16) float biasL[4 * 1024];      // biases
    __shared__ alignas(16) float woutL[8 * 128];       // output projection weights
    __shared__            float boutL[8];

    const int tid   = threadIdx.x;
    const int w     = tid >> 5;
    const int lane  = tid & 31;
    const int brow0 = blockIdx.x * 16;

    // one-time LDS setup
    {
        bf16* hc = &hcur[0][0];
        #pragma unroll
        for (int i = 0; i < 4; ++i)
            *(u32x4*)(hc + tid * 8 + i * 2048) = (u32x4){0u, 0u, 0u, 0u};
        #pragma unroll
        for (int l = 0; l < 4; ++l)
            for (int i = tid; i < 1024; i += 256) biasL[l * 1024 + i] = p.b[l][i];
        for (int i = tid; i < 1024; i += 256) woutL[i] = p.Wout[i];
        if (tid < 8) boutL[tid] = p.bout[tid];
    }
    __syncthreads();

    for (int t = 0; t < 256; ++t) {
        const float* xin = p.x + ((size_t)t * 1024 + brow0) * 64;
        // layer 0 input = x_t (f32 -> bf16, 4 elems/thread)
        {
            const int m = tid >> 4, k = (tid & 15) * 4;
            f32x4 v = *(const f32x4*)(xin + (size_t)m * 64 + k);
            H4 u; u.h[0] = (bf16)v[0]; u.h[1] = (bf16)v[1]; u.h[2] = (bf16)v[2]; u.h[3] = (bf16)v[3];
            *(unsigned long long*)(xh + m * XH_STRIDE + k) = u.q;
        }
        layer_step<0, 320,  1>(t, w, lane, brow0, p.Wpk[0], biasL + 0 * 1024, p.bufH[0], p.bufC[0], xh, hcur[0], outF);
        layer_step<1, 384,  3>(t, w, lane, brow0, p.Wpk[1], biasL + 1 * 1024, p.bufH[1], p.bufC[1], xh, hcur[1], outF);
        // layer 2 input = [out1 | x_t]
        {
            const int m = tid >> 4, k = (tid & 15) * 4;
            f32x4 v = *(const f32x4*)(xin + (size_t)m * 64 + k);
            H4 u; u.h[0] = (bf16)v[0]; u.h[1] = (bf16)v[1]; u.h[2] = (bf16)v[2]; u.h[3] = (bf16)v[3];
            *(unsigned long long*)(xh + m * XH_STRIDE + 128 + k) = u.q;
        }
        layer_step<2, 448,  6>(t, w, lane, brow0, p.Wpk[2], biasL + 2 * 1024, p.bufH[2], p.bufC[2], xh, hcur[2], outF);
        layer_step<3, 384, 12>(t, w, lane, brow0, p.Wpk[3], biasL + 3 * 1024, p.bufH[3], p.bufC[3], xh, hcur[3], outF);

        // y(16x8) = outF(16x128) @ Wout^T + bout
        if (tid < 128) {
            const int m = tid >> 3, o = tid & 7;
            const float* wr = woutL + o * 128;
            float s = boutL[o];
            #pragma unroll 4
            for (int k = 0; k < 128; ++k) s += outF[m * 128 + k] * wr[k];
            p.out[((size_t)t * 1024 + brow0 + m) * 8 + o] = s;
        }
        __syncthreads();
    }
}

extern "C" void kernel_launch(void* const* d_in, const int* in_sizes, int n_in,
                              void* d_out, int out_size, void* d_ws, size_t ws_size,
                              hipStream_t stream)
{
    (void)in_sizes; (void)n_in; (void)out_size; (void)ws_size;

    const float* W[4] = { (const float*)d_in[1], (const float*)d_in[3],
                          (const float*)d_in[5], (const float*)d_in[7] };

    static const int FANS[4] = { 320, 384, 448, 384 };
    static const int DILS[4] = { 1, 3, 6, 12 };

    char*  ws  = (char*)d_ws;
    size_t off = 0;
    auto carve = [&](size_t bytes) -> void* {
        void* ptr = (void*)(ws + off);
        off += (bytes + 255) & ~(size_t)255;
        return ptr;
    };

    Params p;
    p.x    = (const float*)d_in[0];
    p.Wout = (const float*)d_in[9];
    p.bout = (const float*)d_in[10];
    p.out  = (float*)d_out;
    p.b[0] = (const float*)d_in[2];  p.b[1] = (const float*)d_in[4];
    p.b[2] = (const float*)d_in[6];  p.b[3] = (const float*)d_in[8];
    for (int i = 0; i < 4; ++i) p.Wpk[i]  = (const bf16*)carve((size_t)1024 * FANS[i] * sizeof(bf16));
    for (int i = 0; i < 4; ++i) p.bufH[i] = (bf16*)carve((size_t)DILS[i] * 1024 * 128 * sizeof(bf16));
    for (int i = 0; i < 4; ++i) p.bufC[i] = (float*)carve((size_t)DILS[i] * 1024 * 256 * sizeof(float));

    for (int i = 0; i < 4; ++i) {
        int nk    = FANS[i] / 32;
        int total = 1024 * FANS[i];
        dsrnn_pack<<<(total + 255) / 256, 256, 0, stream>>>(W[i], (bf16*)p.Wpk[i], FANS[i], nk);
    }

    dsrnn_main<<<64, 256, 0, stream>>>(p);
}
